// HeteroRGINLayer_49606872269197
// MI455X (gfx1250) — compile-verified, matrix operand order checked by reading, not run on hardware
//
#include <hip/hip_runtime.h>

typedef __attribute__((ext_vector_type(16))) __bf16 v16bf;
typedef __attribute__((ext_vector_type(8)))  __bf16 v8bf;
typedef __attribute__((ext_vector_type(8)))  float  v8f;
typedef __attribute__((ext_vector_type(4)))  float  v4f;

#define N_NODES 100000
#define N_EDGES 500000
#define D 256

// ---------------------------------------------------------------------------
// 1) x (f32) -> x_bf16, 8 elems per thread
// ---------------------------------------------------------------------------
__global__ void convert_x_kernel(const float* __restrict__ x,
                                 __bf16* __restrict__ xb, int n) {
    int i = (blockIdx.x * blockDim.x + threadIdx.x) * 8;
    if (i >= n) return;
    v4f a = *(const v4f*)(x + i);
    v4f b = *(const v4f*)(x + i + 4);
    v8bf o;
#pragma unroll
    for (int j = 0; j < 4; ++j) {
        o[j]     = (__bf16)a[j];
        o[4 + j] = (__bf16)b[j];
    }
    *(v8bf*)(xb + i) = o;
}

// ---------------------------------------------------------------------------
// 2) W (f32, KxN row-major) -> Wt_bf16 (NxK row-major), all three weights
// ---------------------------------------------------------------------------
__global__ void convert_w_kernel(const float* __restrict__ Ws,
                                 const float* __restrict__ W1,
                                 const float* __restrict__ W2,
                                 __bf16* __restrict__ wtS,
                                 __bf16* __restrict__ wt1,
                                 __bf16* __restrict__ wt2) {
    int t = blockIdx.x * blockDim.x + threadIdx.x;   // 0 .. 65535
    int k = t >> 8;
    int n = t & 255;
    int o = n * D + k;                               // transposed position
    wtS[o] = (__bf16)Ws[t];
    wt1[o] = (__bf16)W1[t];
    wt2[o] = (__bf16)W2[t];
}

// ---------------------------------------------------------------------------
// 3) Fused triple GEMM: one wave computes a 16-row output panel for all three
//    weights using v_wmma_f32_16x16x32_bf16. A panel (16x256 bf16) is loaded
//    once into registers; B tiles come from the pre-transposed bf16 weights.
// ---------------------------------------------------------------------------
__global__ __launch_bounds__(256)
void rgin_gemm3_kernel(const __bf16* __restrict__ xb,
                       const __bf16* __restrict__ wtS,
                       const __bf16* __restrict__ wt1,
                       const __bf16* __restrict__ wt2,
                       float*  __restrict__ h,
                       __bf16* __restrict__ msg1,
                       __bf16* __restrict__ msg2) {
    const int lane = threadIdx.x & 31;
    const int wave = (blockIdx.x * blockDim.x + threadIdx.x) >> 5;
    const int row0 = wave * 16;
    if (row0 >= N_NODES) return;            // wave-uniform: EXEC stays all-1s

    const int r  = lane & 15;               // M (for A) / N (for B) within tile
    const int k0 = (lane >> 4) * 8;         // K sub-offset from lane half

    // --- load A panel: ISA 16-bit A layout, 8 chunks of K=32 ---
    const __bf16* arow = xb + (size_t)(row0 + r) * D;
    v16bf A[8];
#pragma unroll
    for (int kc = 0; kc < 8; ++kc) {
        v8bf lo = *(const v8bf*)(arow + kc * 32 + k0);        // K = 32kc+k0..+7
        v8bf hi = *(const v8bf*)(arow + kc * 32 + 16 + k0);   // K = +16
#pragma unroll
        for (int i = 0; i < 8; ++i) { A[kc][i] = lo[i]; A[kc][8 + i] = hi[i]; }
    }

    const __bf16* wts[3] = { wtS, wt1, wt2 };
    const int m0  = (lane >> 4) * 8;        // C layout: M = j + 8*(lane/16)

    for (int w = 0; w < 3; ++w) {
        const __bf16* wt = wts[w];
#pragma unroll 1
        for (int nt = 0; nt < 16; ++nt) {
            // B element (k, n) = Wt[n][k]; lane owns column n = nt*16 + r
            const __bf16* bcol = wt + (size_t)(nt * 16 + r) * D + k0;
            v8f acc = {};
#pragma unroll
            for (int kc = 0; kc < 8; ++kc) {
                v8bf lo = *(const v8bf*)(bcol + kc * 32);
                v8bf hi = *(const v8bf*)(bcol + kc * 32 + 16);
                v16bf B;
#pragma unroll
                for (int i = 0; i < 8; ++i) { B[i] = lo[i]; B[8 + i] = hi[i]; }
                acc = __builtin_amdgcn_wmma_f32_16x16x32_bf16(
                          false, A[kc], false, B, (short)0, acc, false, false);
            }
            const int col = nt * 16 + r;
            if (w == 0) {
#pragma unroll
                for (int j = 0; j < 8; ++j)
                    h[(size_t)(row0 + m0 + j) * D + col] = 1.1f * acc[j];
            } else {
                __bf16* mo = (w == 1) ? msg1 : msg2;
#pragma unroll
                for (int j = 0; j < 8; ++j)
                    mo[(size_t)(row0 + m0 + j) * D + col] = (__bf16)acc[j];
            }
        }
    }
}

// ---------------------------------------------------------------------------
// 4) Edge scatter: one wave per edge; lane grabs 8 bf16 feats (one b128 load)
//    and does 8 relaxed agent-scope f32 atomic adds (L2-resident accumulator).
// ---------------------------------------------------------------------------
__global__ __launch_bounds__(256)
void scatter_add_kernel(const __bf16* __restrict__ msg,
                        const int* __restrict__ src,
                        const int* __restrict__ dst,
                        float* __restrict__ h, int n_edges) {
    const int wave = (blockIdx.x * blockDim.x + threadIdx.x) >> 5;
    const int lane = threadIdx.x & 31;
    if (wave >= n_edges) return;
    const int s = src[wave];
    const int d = dst[wave];
    const v8bf m = *(const v8bf*)(msg + (size_t)s * D + lane * 8);
    float* hp = h + (size_t)d * D + lane * 8;
#pragma unroll
    for (int i = 0; i < 8; ++i)
        __hip_atomic_fetch_add(hp + i, (float)m[i],
                               __ATOMIC_RELAXED, __HIP_MEMORY_SCOPE_AGENT);
}

// ---------------------------------------------------------------------------
// 5) Epilogue: h = relu(h + bias)
// ---------------------------------------------------------------------------
__global__ void bias_relu_kernel(float* __restrict__ h,
                                 const float* __restrict__ bias, int n) {
    int i = (blockIdx.x * blockDim.x + threadIdx.x) * 4;
    if (i >= n) return;
    v4f v = *(v4f*)(h + i);
    v4f b = *(const v4f*)(bias + (i & (D - 1)));
#pragma unroll
    for (int j = 0; j < 4; ++j) v[j] = fmaxf(v[j] + b[j], 0.0f);
    *(v4f*)(h + i) = v;
}

// ---------------------------------------------------------------------------
extern "C" void kernel_launch(void* const* d_in, const int* in_sizes, int n_in,
                              void* d_out, int out_size, void* d_ws, size_t ws_size,
                              hipStream_t stream) {
    const float* x    = (const float*)d_in[0];
    const float* Ws   = (const float*)d_in[1];
    const float* W1   = (const float*)d_in[2];
    const float* W2   = (const float*)d_in[3];
    const float* bias = (const float*)d_in[4];
    const int* src1   = (const int*)d_in[5];
    const int* dst1   = (const int*)d_in[6];
    const int* src2   = (const int*)d_in[7];
    const int* dst2   = (const int*)d_in[8];
    float* h = (float*)d_out;

    // workspace layout (all 16B aligned)
    char* ws = (char*)d_ws;
    size_t off = 0;
    __bf16* xb   = (__bf16*)(ws + off); off += (size_t)N_NODES * D * 2;  // 51.2 MB
    __bf16* wtS  = (__bf16*)(ws + off); off += (size_t)D * D * 2;
    __bf16* wt1  = (__bf16*)(ws + off); off += (size_t)D * D * 2;
    __bf16* wt2  = (__bf16*)(ws + off); off += (size_t)D * D * 2;
    __bf16* msg1 = (__bf16*)(ws + off); off += (size_t)N_NODES * D * 2;  // 51.2 MB
    __bf16* msg2 = (__bf16*)(ws + off); off += (size_t)N_NODES * D * 2;  // 51.2 MB

    const int nelem = N_NODES * D;  // 25,600,000

    convert_x_kernel<<<(nelem / 8 + 255) / 256, 256, 0, stream>>>(x, xb, nelem);
    convert_w_kernel<<<(D * D) / 256, 256, 0, stream>>>(Ws, W1, W2, wtS, wt1, wt2);

    const int n_waves  = N_NODES / 16;                  // 6250 row panels
    const int n_blocks = (n_waves + 7) / 8;             // 8 waves / block
    rgin_gemm3_kernel<<<n_blocks, 256, 0, stream>>>(xb, wtS, wt1, wt2, h, msg1, msg2);

    const int sc_blocks = (N_EDGES + 7) / 8;            // 8 edges / block
    scatter_add_kernel<<<sc_blocks, 256, 0, stream>>>(msg1, src1, dst1, h, N_EDGES);
    scatter_add_kernel<<<sc_blocks, 256, 0, stream>>>(msg2, src2, dst2, h, N_EDGES);

    bias_relu_kernel<<<(nelem / 4 + 255) / 256, 256, 0, stream>>>(h, bias, nelem);
}
